// TransformerEncoderLayer_50328426774895
// MI455X (gfx1250) — matrix operand order repaired and checked by reading, not compiled, not run on hardware
//
#include <hip/hip_runtime.h>
#include <math.h>

#define BATCH  4
#define SEQ    2048
#define DMODEL 512
#define NHEAD  8
#define HDIM   64
#define DFF    2048
#define MROWS  (BATCH*SEQ)   // 8192

typedef __bf16 bf16_t;
typedef bf16_t bf16x8  __attribute__((ext_vector_type(8)));
typedef bf16_t bf16x16 __attribute__((ext_vector_type(16)));
typedef float  f32x8   __attribute__((ext_vector_type(8)));

static __device__ inline f32x8 wmma_bf16(bf16x16 a, bf16x16 b, f32x8 c) {
  return __builtin_amdgcn_wmma_f32_16x16x32_bf16(false, a, false, b, (short)0, c, false, false);
}

static __device__ inline bf16x16 cat8(bf16x8 lo, bf16x8 hi) {
  return __builtin_shufflevector(lo, hi, 0,1,2,3,4,5,6,7,8,9,10,11,12,13,14,15);
}

// A fragment (16x32 bf16, M x K): lane = row (lane&15); per-lane two contiguous
// 8-element K chunks at kbase + 8*(lane>=16) and +16 more.
static __device__ inline bf16x16 load_a_frag(const bf16_t* A, int lda, int rowbase,
                                             int kbase, int lane) {
  int row  = rowbase + (lane & 15);
  int base = (lane >> 4) * 8;
  const bf16_t* p = A + (size_t)row * lda + kbase + base;
  bf16x8 lo = *(const bf16x8*)p;
  bf16x8 hi = *(const bf16x8*)(p + 16);
  return cat8(lo, hi);
}

// B fragment (32x16 bf16, K x N) from K-major matrix Bt[k][n]:
// lane = column (lane&15); per-lane K at kbase + 8*(lane>=16) + s and +16 more.
static __device__ inline bf16x16 load_b_frag(const bf16_t* Bt, int ldb, int kbase,
                                             int nbase, int lane) {
  const bf16_t* p = Bt + (size_t)(kbase + (lane >> 4) * 8) * ldb + nbase + (lane & 15);
  bf16x16 r;
  #pragma unroll
  for (int s = 0; s < 8; s++) {
    r[s]     = p[(size_t)s * ldb];
    r[8 + s] = p[(size_t)(16 + s) * ldb];
  }
  return r;
}

// ---------------------------------------------------------------- pack kernels
__global__ void cvt_bf16_kernel(const float* __restrict__ in, bf16_t* __restrict__ out, int n) {
  int i = blockIdx.x * blockDim.x + threadIdx.x;
  if (i < n) out[i] = (bf16_t)in[i];
}

// dst[k*dstLd + colOff + n] = W[n*nIn + k]   (W is (nOut, nIn) row-major fp32)
__global__ void transpose_w_kernel(const float* __restrict__ W, bf16_t* __restrict__ dst,
                                   int nOut, int nIn, int dstLd, int colOff) {
  int i = blockIdx.x * blockDim.x + threadIdx.x;
  if (i >= nOut * nIn) return;
  int k = i / nOut;
  int n = i - k * nOut;
  dst[(size_t)k * dstLd + colOff + n] = (bf16_t)W[(size_t)n * nIn + k];
}

// ------------------------------------------------------------- fused QKV GEMM
// Y = srcb(M x 512) @ wqkvT(512 x 1536), scatter into Q (b,h,t,d),
// Kt (b,h,d,t), V (b,h,t,d) with per-segment bias.
__global__ __launch_bounds__(256) void qkv_gemm_kernel(
    const bf16_t* __restrict__ A, const bf16_t* __restrict__ Bt,
    const float* __restrict__ bq, const float* __restrict__ bk, const float* __restrict__ bv,
    bf16_t* __restrict__ Q, bf16_t* __restrict__ Kt, bf16_t* __restrict__ V) {
  const int K = DMODEL, N = 3 * DMODEL;
  int lane = threadIdx.x & 31, wid = threadIdx.x >> 5;
  int mb = blockIdx.x * 128 + (wid >> 1) * 32;
  int nb = blockIdx.y * 128 + (wid & 1) * 64;
  f32x8 acc[2][4] = {};
  for (int k = 0; k < K; k += 32) {
    bf16x16 a0 = load_a_frag(A, K, mb,      k, lane);
    bf16x16 a1 = load_a_frag(A, K, mb + 16, k, lane);
    #pragma unroll
    for (int n = 0; n < 4; n++) {
      bf16x16 bfr = load_b_frag(Bt, N, k, nb + n * 16, lane);
      acc[0][n] = wmma_bf16(a0, bfr, acc[0][n]);
      acc[1][n] = wmma_bf16(a1, bfr, acc[1][n]);
    }
  }
  int rl = lane & 15, hf = lane >> 4;
  #pragma unroll
  for (int mi = 0; mi < 2; mi++)
    #pragma unroll
    for (int n = 0; n < 4; n++)
      #pragma unroll
      for (int j = 0; j < 8; j++) {
        int m  = mb + mi * 16 + j + hf * 8;
        int ng = nb + n * 16 + rl;
        int b = m >> 11, t = m & (SEQ - 1);
        int seg = ng >> 9, nn = ng & 511;
        int h = nn >> 6, d = nn & 63;
        float bias = (seg == 0) ? bq[nn] : (seg == 1) ? bk[nn] : bv[nn];
        float v = acc[mi][n][j] + bias;
        size_t bh = (size_t)(b * NHEAD + h);
        if (seg == 0)      Q [(bh * SEQ  + t) * HDIM + d] = (bf16_t)v;
        else if (seg == 1) Kt[(bh * HDIM + d) * SEQ  + t] = (bf16_t)v;
        else               V [(bh * SEQ  + t) * HDIM + d] = (bf16_t)v;
      }
}

// ------------------------------------------------------------ flash attention
// Block = 4 waves, each wave owns 16 query rows of one (b,h); streams keys in
// blocks of 32 with online softmax; O accumulated with WMMA over P*V.
__global__ __launch_bounds__(128) void flash_attn_kernel(
    const bf16_t* __restrict__ Q, const bf16_t* __restrict__ Kt,
    const bf16_t* __restrict__ V, bf16_t* __restrict__ ctx) {
  __shared__ alignas(16) bf16_t plds[4][16 * 32];
  int lane = threadIdx.x & 31, wid = threadIdx.x >> 5;
  int bh = blockIdx.x >> 5;        // 32 query-blocks per (b,h)
  int qblk = blockIdx.x & 31;
  int b = bh >> 3, h = bh & 7;
  int qbase = qblk * 64 + wid * 16;
  const bf16_t* Qp = Q  + (size_t)bh * SEQ  * HDIM;
  const bf16_t* Kp = Kt + (size_t)bh * HDIM * SEQ;
  const bf16_t* Vp = V  + (size_t)bh * SEQ  * HDIM;

  bf16x16 qf0 = load_a_frag(Qp, HDIM, qbase, 0,  lane);
  bf16x16 qf1 = load_a_frag(Qp, HDIM, qbase, 32, lane);

  f32x8 o[4] = {};
  float mrow[8], lrow[8];
  #pragma unroll
  for (int j = 0; j < 8; j++) { mrow[j] = -3.0e38f; lrow[j] = 0.f; }

  int rl = lane & 15, hf = lane >> 4;
  const float scale = 0.125f;  // 1/sqrt(64)

  for (int kp = 0; kp < SEQ; kp += 32) {
    f32x8 s0 = {}, s1 = {};
    s0 = wmma_bf16(qf0, load_b_frag(Kp, SEQ, 0,  kp,      lane), s0);
    s0 = wmma_bf16(qf1, load_b_frag(Kp, SEQ, 32, kp,      lane), s0);
    s1 = wmma_bf16(qf0, load_b_frag(Kp, SEQ, 0,  kp + 16, lane), s1);
    s1 = wmma_bf16(qf1, load_b_frag(Kp, SEQ, 32, kp + 16, lane), s1);

    #pragma unroll
    for (int j = 0; j < 8; j++) {
      float a = s0[j] * scale, c = s1[j] * scale;
      float mx = fmaxf(a, c);
      for (int msk = 1; msk < 16; msk <<= 1) mx = fmaxf(mx, __shfl_xor(mx, msk, 32));
      float mnew = fmaxf(mrow[j], mx);
      float corr = __expf(mrow[j] - mnew);
      float p0 = __expf(a - mnew);
      float p1 = __expf(c - mnew);
      float rs = p0 + p1;
      for (int msk = 1; msk < 16; msk <<= 1) rs += __shfl_xor(rs, msk, 32);
      lrow[j] = lrow[j] * corr + rs;
      mrow[j] = mnew;
      #pragma unroll
      for (int n = 0; n < 4; n++) o[n][j] *= corr;
      int row = j + 8 * hf;                     // C-layout -> row-major P in LDS
      plds[wid][row * 32 + rl]      = (bf16_t)p0;
      plds[wid][row * 32 + 16 + rl] = (bf16_t)p1;
    }
    asm volatile("s_wait_dscnt 0x0" ::: "memory");  // per-wave LDS RAW fence
    {
      int row = rl, base = hf * 8;              // re-read P in A-fragment layout
      bf16x8 lo = *(const bf16x8*)&plds[wid][row * 32 + base];
      bf16x8 hi = *(const bf16x8*)&plds[wid][row * 32 + 16 + base];
      bf16x16 pf = cat8(lo, hi);
      #pragma unroll
      for (int n = 0; n < 4; n++)
        o[n] = wmma_bf16(pf, load_b_frag(Vp, HDIM, kp, n * 16, lane), o[n]);
    }
  }

  #pragma unroll
  for (int n = 0; n < 4; n++)
    #pragma unroll
    for (int j = 0; j < 8; j++) {
      int t = qbase + j + 8 * hf;
      int d = n * 16 + rl;
      float v = o[n][j] / lrow[j];
      ctx[((size_t)b * SEQ + t) * DMODEL + h * HDIM + d] = (bf16_t)v;
    }
}

// ------------------------------------------------------------- generic GEMM
// mode 0: outF = A@Bt + bias (fp32).   mode 1: outB = GELU(A@Bt + bias) (bf16).
__global__ __launch_bounds__(256) void gemm_kernel(
    const bf16_t* __restrict__ A, const bf16_t* __restrict__ Bt,
    const float* __restrict__ bias, float* __restrict__ outF,
    bf16_t* __restrict__ outB, int N, int K, int mode) {
  int lane = threadIdx.x & 31, wid = threadIdx.x >> 5;
  int mb = blockIdx.x * 128 + (wid >> 1) * 32;
  int nb = blockIdx.y * 128 + (wid & 1) * 64;
  f32x8 acc[2][4] = {};
  for (int k = 0; k < K; k += 32) {
    bf16x16 a0 = load_a_frag(A, K, mb,      k, lane);
    bf16x16 a1 = load_a_frag(A, K, mb + 16, k, lane);
    #pragma unroll
    for (int n = 0; n < 4; n++) {
      bf16x16 bfr = load_b_frag(Bt, N, k, nb + n * 16, lane);
      acc[0][n] = wmma_bf16(a0, bfr, acc[0][n]);
      acc[1][n] = wmma_bf16(a1, bfr, acc[1][n]);
    }
  }
  int rl = lane & 15, hf = lane >> 4;
  #pragma unroll
  for (int mi = 0; mi < 2; mi++)
    #pragma unroll
    for (int n = 0; n < 4; n++)
      #pragma unroll
      for (int j = 0; j < 8; j++) {
        int m  = mb + mi * 16 + j + hf * 8;
        int ng = nb + n * 16 + rl;
        float v = acc[mi][n][j] + bias[ng];
        if (mode == 0) {
          outF[(size_t)m * N + ng] = v;
        } else {
          float gl = 0.5f * v * (1.0f + erff(v * 0.70710678118654752f));
          outB[(size_t)m * N + ng] = (bf16_t)gl;
        }
      }
}

// ------------------------------------------------ residual + LayerNorm (fp32)
__global__ __launch_bounds__(256) void add_ln_kernel(
    const float* __restrict__ xa, const float* __restrict__ xb,
    const float* __restrict__ g, const float* __restrict__ be,
    float* __restrict__ outF, bf16_t* __restrict__ outB) {
  int lane = threadIdx.x & 31, wid = threadIdx.x >> 5;
  int row = blockIdx.x * 8 + wid;
  const float* pa = xa + (size_t)row * DMODEL + lane * 16;
  const float* pb = xb + (size_t)row * DMODEL + lane * 16;
  float v[16];
  float s = 0.f;
  #pragma unroll
  for (int i = 0; i < 16; i++) { v[i] = pa[i] + pb[i]; s += v[i]; }
  for (int msk = 1; msk < 32; msk <<= 1) s += __shfl_xor(s, msk, 32);
  float mu = s * (1.0f / DMODEL);
  float q = 0.f;
  #pragma unroll
  for (int i = 0; i < 16; i++) { float d = v[i] - mu; q += d * d; }
  for (int msk = 1; msk < 32; msk <<= 1) q += __shfl_xor(q, msk, 32);
  float rstd = rsqrtf(q * (1.0f / DMODEL) + 1e-5f);
  #pragma unroll
  for (int i = 0; i < 16; i++) {
    int c = lane * 16 + i;
    float y = (v[i] - mu) * rstd * g[c] + be[c];
    outF[(size_t)row * DMODEL + c] = y;
    if (outB) outB[(size_t)row * DMODEL + c] = (bf16_t)y;
  }
}

// ----------------------------------------------------------------- launcher
extern "C" void kernel_launch(void* const* d_in, const int* in_sizes, int n_in,
                              void* d_out, int out_size, void* d_ws, size_t ws_size,
                              hipStream_t stream) {
  (void)in_sizes; (void)n_in; (void)out_size; (void)ws_size;
  const float* src = (const float*)d_in[0];
  const float* Wq  = (const float*)d_in[1];
  const float* bq  = (const float*)d_in[2];
  const float* Wk  = (const float*)d_in[3];
  const float* bk  = (const float*)d_in[4];
  const float* Wv  = (const float*)d_in[5];
  const float* bv  = (const float*)d_in[6];
  const float* Wo  = (const float*)d_in[7];
  const float* bo  = (const float*)d_in[8];
  const float* W1  = (const float*)d_in[9];
  const float* b1  = (const float*)d_in[10];
  const float* W2  = (const float*)d_in[11];
  const float* b2  = (const float*)d_in[12];
  const float* g1  = (const float*)d_in[13];
  const float* be1 = (const float*)d_in[14];
  const float* g2  = (const float*)d_in[15];
  const float* be2 = (const float*)d_in[16];
  float* out = (float*)d_out;

  char* w = (char*)d_ws;
  auto carve = [&](size_t bytes) -> char* {
    char* p = w; w += (bytes + 255) & ~(size_t)255; return p;
  };
  bf16_t* srcb  = (bf16_t*)carve((size_t)MROWS * DMODEL * 2);
  bf16_t* wqkvT = (bf16_t*)carve((size_t)DMODEL * 3 * DMODEL * 2);
  bf16_t* woT   = (bf16_t*)carve((size_t)DMODEL * DMODEL * 2);
  bf16_t* w1T   = (bf16_t*)carve((size_t)DMODEL * DFF * 2);
  bf16_t* w2T   = (bf16_t*)carve((size_t)DFF * DMODEL * 2);
  bf16_t* Qb    = (bf16_t*)carve((size_t)MROWS * DMODEL * 2);
  bf16_t* Ktb   = (bf16_t*)carve((size_t)MROWS * DMODEL * 2);
  bf16_t* Vb    = (bf16_t*)carve((size_t)MROWS * DMODEL * 2);
  bf16_t* ctx   = (bf16_t*)carve((size_t)MROWS * DMODEL * 2);
  float*  attn  = (float*) carve((size_t)MROWS * DMODEL * 4);
  float*  x     = (float*) carve((size_t)MROWS * DMODEL * 4);
  bf16_t* xbb   = (bf16_t*)carve((size_t)MROWS * DMODEL * 2);
  bf16_t* hb    = (bf16_t*)carve((size_t)MROWS * DFF * 2);
  float*  ffn   = (float*) carve((size_t)MROWS * DMODEL * 4);

  {
    int n = MROWS * DMODEL;
    cvt_bf16_kernel<<<(n + 255) / 256, 256, 0, stream>>>(src, srcb, n);
  }
  int nw = DMODEL * DMODEL;
  transpose_w_kernel<<<(nw + 255) / 256, 256, 0, stream>>>(Wq, wqkvT, DMODEL, DMODEL, 3 * DMODEL, 0);
  transpose_w_kernel<<<(nw + 255) / 256, 256, 0, stream>>>(Wk, wqkvT, DMODEL, DMODEL, 3 * DMODEL, DMODEL);
  transpose_w_kernel<<<(nw + 255) / 256, 256, 0, stream>>>(Wv, wqkvT, DMODEL, DMODEL, 3 * DMODEL, 2 * DMODEL);
  transpose_w_kernel<<<(nw + 255) / 256, 256, 0, stream>>>(Wo, woT, DMODEL, DMODEL, DMODEL, 0);
  int nf = DFF * DMODEL;
  transpose_w_kernel<<<(nf + 255) / 256, 256, 0, stream>>>(W1, w1T, DFF, DMODEL, DFF, 0);
  transpose_w_kernel<<<(nf + 255) / 256, 256, 0, stream>>>(W2, w2T, DMODEL, DFF, DMODEL, 0);

  qkv_gemm_kernel<<<dim3(MROWS / 128, (3 * DMODEL) / 128), 256, 0, stream>>>(
      srcb, wqkvT, bq, bk, bv, Qb, Ktb, Vb);
  flash_attn_kernel<<<BATCH * NHEAD * (SEQ / 64), 128, 0, stream>>>(Qb, Ktb, Vb, ctx);
  gemm_kernel<<<dim3(MROWS / 128, DMODEL / 128), 256, 0, stream>>>(
      ctx, woT, bo, attn, nullptr, DMODEL, DMODEL, 0);
  add_ln_kernel<<<MROWS / 8, 256, 0, stream>>>(src, attn, g1, be1, x, xbb);
  gemm_kernel<<<dim3(MROWS / 128, DFF / 128), 256, 0, stream>>>(
      xbb, w1T, b1, nullptr, hb, DFF, DMODEL, 1);
  gemm_kernel<<<dim3(MROWS / 128, DMODEL / 128), 256, 0, stream>>>(
      hb, w2T, b2, ffn, nullptr, DMODEL, DFF, 0);
  add_ln_kernel<<<MROWS / 8, 256, 0, stream>>>(x, ffn, g2, be2, out, nullptr);
}
